// Transformer_Decoder_42142219108894
// MI455X (gfx1250) — matrix operand order repaired and checked by reading, not compiled, not run on hardware
//
#include <hip/hip_runtime.h>

// ---------------------------------------------------------------------------
// CDNA5 / gfx1250 transformer decoder block.
// All GEMMs: bf16 operands (pre-converted in global), v_wmma_f32_16x16x32_bf16,
// LDS tiles staged with GLOBAL_LOAD_ASYNC_TO_LDS_B128 (double-buffered) when
// the toolchain exposes the builtin; uint4 copy fallback otherwise.
// Uniform GEMM convention: A bf16 [M,K] row-major, B bf16 [N,K] row-major
// (k-contiguous), so every 16B global chunk maps to a contiguous 16B LDS
// fragment chunk.
// ---------------------------------------------------------------------------

typedef __bf16 v16bf __attribute__((ext_vector_type(16)));
typedef float  v8f   __attribute__((ext_vector_type(8)));
typedef int    v4i   __attribute__((ext_vector_type(4)));
typedef unsigned int uint32;

#if defined(__has_builtin)
#  if __has_builtin(__builtin_amdgcn_global_load_async_to_lds_b128)
#    define USE_ASYNC_LDS 1
#  endif
#endif
#ifndef USE_ASYNC_LDS
#  define USE_ASYNC_LDS 0
#endif

__device__ __forceinline__ unsigned short f2bf(float f) {
    // round-to-nearest-even fp32 -> bf16
    unsigned int u = __float_as_uint(f);
    u += 0x7FFFu + ((u >> 16) & 1u);
    return (unsigned short)(u >> 16);
}
__device__ __forceinline__ uint32 pack2(float a, float b) {
    return (uint32)f2bf(a) | ((uint32)f2bf(b) << 16);
}

// per-lane 16-byte global -> LDS copy
__device__ __forceinline__ void copy16_g2l(const unsigned short* g, unsigned short* l) {
#if USE_ASYNC_LDS
    // signature (from clang diagnostic): (int4 AS1*, int4 AS3*, i32, i32)
    __attribute__((address_space(1))) v4i* gp =
        (__attribute__((address_space(1))) v4i*)((void*)g);
    __attribute__((address_space(3))) v4i* lp =
        (__attribute__((address_space(3))) v4i*)((void*)l);
    __builtin_amdgcn_global_load_async_to_lds_b128(gp, lp, 0, 0);
#else
    *(uint4*)l = *(const uint4*)g;
#endif
}
__device__ __forceinline__ void staging_wait() {
#if USE_ASYNC_LDS
#  if __has_builtin(__builtin_amdgcn_s_wait_asynccnt)
    __builtin_amdgcn_s_wait_asynccnt(0);
#  else
    asm volatile("s_wait_asynccnt 0x0" ::: "memory");
#  endif
#endif
}

// ---------------------------------------------------------------------------
// GEMM: C = epi( scale * A x B^T )  with A bf16 [M,K] lda, B bf16 [N,K] ldb.
// OUT: 0 = f32 C [M,Ncols] ldc ; 1 = bf16 C^T [Ncols,M] ldc(=M) ;
//      2 = bf16 C [M,Ncols] ldc.  RELU applied after scale.
// Block tile 128x128, K-step 32, 256 thr = 8 waves, wave = 2x4 WMMA tiles.
// LDS fragment order (ISA 7.12.2):
//   A: lane = mLoc + 16*((k>>3)&1), e = (k&7) + 8*((k>>4)&1)
//   B: lane = nLoc + 16*((k>>3)&1), e = (k&7) + 8*((k>>4)&1)
// => A chunk ch(8k): grp=ch&1, e0=8*(ch>>1); B chunk ch(8k): grp=ch&1, e0=8*(ch>>1)
// ---------------------------------------------------------------------------
template<int OUT, int RELU>
__global__ __launch_bounds__(256)
void gemm_bf16_wmma(const unsigned short* __restrict__ Ag,
                    const unsigned short* __restrict__ Bg,
                    void* __restrict__ Cv,
                    int M, int Ncols, int K,
                    int lda, int ldb, int ldc,
                    long long strideA, long long strideB, long long strideC,
                    float scale)
{
    __shared__ __align__(16) unsigned short shA[2][8 * 32 * 16]; // 2 x 8KB
    __shared__ __align__(16) unsigned short shB[2][8 * 32 * 16]; // 2 x 8KB

    Ag += (long long)blockIdx.z * strideA;
    Bg += (long long)blockIdx.z * strideB;

    const int m0   = blockIdx.x * 128;
    const int n0   = blockIdx.y * 128;
    const int tid  = threadIdx.x;
    const int lane = tid & 31;
    const int wv   = tid >> 5;
    const int mtBase = (wv >> 1) * 2;
    const int ntBase = (wv & 1) * 4;

    auto stage = [&](int k0, int buf) {
        #pragma unroll
        for (int c = 0; c < 2; ++c) {             // A: 512 x 16B chunks
            int it = tid + 256 * c;
            int m = it >> 2, ch = it & 3;
            const unsigned short* g = Ag + (long long)(m0 + m) * lda + (k0 + ch * 8);
            unsigned short* l = &shA[buf][((m >> 4) * 32 + (m & 15) + 16 * (ch & 1)) * 16
                                          + 8 * (ch >> 1)];
            copy16_g2l(g, l);
        }
        #pragma unroll
        for (int c = 0; c < 2; ++c) {             // B: 512 x 16B chunks
            int it = tid + 256 * c;
            int n = it >> 2, ch = it & 3;
            const unsigned short* g = Bg + (long long)(n0 + n) * ldb + (k0 + ch * 8);
            unsigned short* l = &shB[buf][((n >> 4) * 32 + (n & 15) + 16 * (ch & 1)) * 16
                                          + 8 * (ch >> 1)];
            copy16_g2l(g, l);
        }
    };

    v8f acc[2][4];
    #pragma unroll
    for (int i = 0; i < 2; ++i)
        #pragma unroll
        for (int j = 0; j < 4; ++j)
            #pragma unroll
            for (int r = 0; r < 8; ++r) acc[i][j][r] = 0.0f;

    const int KT = K >> 5;
    stage(0, 0);
    for (int kt = 0; kt < KT; ++kt) {
        staging_wait();
        __syncthreads();
        if (kt + 1 < KT) stage((kt + 1) << 5, (kt + 1) & 1);

        const unsigned short* sa = shA[kt & 1];
        const unsigned short* sb = shB[kt & 1];
        v16bf afrag[2], bfrag[4];
        #pragma unroll
        for (int i = 0; i < 2; ++i)
            afrag[i] = *(const v16bf*)&sa[((mtBase + i) * 32 + lane) * 16];
        #pragma unroll
        for (int j = 0; j < 4; ++j)
            bfrag[j] = *(const v16bf*)&sb[((ntBase + j) * 32 + lane) * 16];

        #pragma unroll
        for (int i = 0; i < 2; ++i)
            #pragma unroll
            for (int j = 0; j < 4; ++j)
                acc[i][j] = __builtin_amdgcn_wmma_f32_16x16x32_bf16(
                    false, afrag[i], false, bfrag[j],
                    (short)0, acc[i][j], false, false);
        // all ds_loads of this buffer retire before the WMMAs issue, which
        // precede next iteration's barrier that gates the overwriting stage
    }

    // ---------------- epilogue (C/D: VGPR r -> M = r + 8*(lane>>4)) --------
    const int colBase = n0 + (lane & 15);
    const int rowHalf = 8 * (lane >> 4);
    #pragma unroll
    for (int i = 0; i < 2; ++i) {
        int row0 = m0 + (mtBase + i) * 16 + rowHalf;
        #pragma unroll
        for (int j = 0; j < 4; ++j) {
            int col = colBase + (ntBase + j) * 16;
            float e[8];
            #pragma unroll
            for (int r = 0; r < 8; ++r) {
                float v = acc[i][j][r] * scale;
                if (RELU) v = fmaxf(v, 0.0f);
                e[r] = v;
            }
            if (OUT == 0) {
                float* Cg = (float*)Cv + (long long)blockIdx.z * strideC;
                #pragma unroll
                for (int r = 0; r < 8; ++r)
                    Cg[(long long)(row0 + r) * ldc + col] = e[r];
            } else if (OUT == 1) {
                // bf16 C^T: 8 consecutive rows -> one 16B store per tile
                unsigned short* Cg = (unsigned short*)Cv + (long long)blockIdx.z * strideC;
                uint4 w;
                w.x = pack2(e[0], e[1]); w.y = pack2(e[2], e[3]);
                w.z = pack2(e[4], e[5]); w.w = pack2(e[6], e[7]);
                *(uint4*)&Cg[(long long)col * ldc + row0] = w;
            } else {
                unsigned short* Cg = (unsigned short*)Cv + (long long)blockIdx.z * strideC;
                #pragma unroll
                for (int r = 0; r < 8; ++r)
                    Cg[(long long)(row0 + r) * ldc + col] = f2bf(e[r]);
            }
        }
    }
}

// ---------------------------------------------------------------------------
// fp32 -> bf16 elementwise (n must be a multiple of 1024; 4 elems/thread)
// ---------------------------------------------------------------------------
__global__ __launch_bounds__(256)
void f32_to_bf16(const float* __restrict__ src, unsigned short* __restrict__ dst)
{
    long long i = ((long long)blockIdx.x * 256 + threadIdx.x) * 4;
    float4 f = *(const float4*)(src + i);
    uint2 w;
    w.x = pack2(f.x, f.y);
    w.y = pack2(f.z, f.w);
    *(uint2*)(dst + i) = w;
}

// ---------------------------------------------------------------------------
// Tiled transpose + convert: src f32 [R,Cc] per batch -> dst bf16 [Cc,R]
// ---------------------------------------------------------------------------
__global__ __launch_bounds__(256)
void transpose_f32_to_bf16(const float* __restrict__ src,
                           unsigned short* __restrict__ dst,
                           int R, int Cc)
{
    __shared__ float tile[32][33];
    const long long ofs = (long long)blockIdx.z * R * Cc;
    const int r0 = blockIdx.x * 32, c0 = blockIdx.y * 32;
    const int tx = threadIdx.x & 31, ty = threadIdx.x >> 5;
    #pragma unroll
    for (int k = 0; k < 4; ++k)
        tile[ty + 8 * k][tx] = src[ofs + (long long)(r0 + ty + 8 * k) * Cc + (c0 + tx)];
    __syncthreads();
    #pragma unroll
    for (int k = 0; k < 4; ++k)
        dst[ofs + (long long)(c0 + ty + 8 * k) * R + (r0 + tx)] = f2bf(tile[tx][ty + 8 * k]);
}

// ---------------------------------------------------------------------------
// Row softmax over L=512: f32 in, bf16 out. One wave per row, 8 rows/block.
// ---------------------------------------------------------------------------
__global__ __launch_bounds__(256)
void softmax_rows_512(const float* __restrict__ src, unsigned short* __restrict__ dst)
{
    const int wv = threadIdx.x >> 5, lane = threadIdx.x & 31;
    const long long row = (long long)blockIdx.x * 8 + wv;
    const float* p = src + row * 512;
    unsigned short* q = dst + row * 512;

    float v[16];
    float mx = -3.402823466e38f;
    #pragma unroll
    for (int i = 0; i < 16; ++i) { v[i] = p[lane + 32 * i]; mx = fmaxf(mx, v[i]); }
    #pragma unroll
    for (int off = 16; off > 0; off >>= 1) mx = fmaxf(mx, __shfl_xor(mx, off, 32));
    float s = 0.0f;
    #pragma unroll
    for (int i = 0; i < 16; ++i) { v[i] = __expf(v[i] - mx); s += v[i]; }
    #pragma unroll
    for (int off = 16; off > 0; off >>= 1) s += __shfl_xor(s, off, 32);
    const float inv = 1.0f / s;
    #pragma unroll
    for (int i = 0; i < 16; ++i) q[lane + 32 * i] = f2bf(v[i] * inv);
}

// ---------------------------------------------------------------------------
// Residual add + LayerNorm over whole [C,N] slab per batch (f32 in/out).
// ---------------------------------------------------------------------------
__global__ __launch_bounds__(256)
void layernorm_add(const float* __restrict__ a, const float* __restrict__ b,
                   const float* __restrict__ gamma, const float* __restrict__ beta,
                   float* __restrict__ out, int n)
{
    __shared__ float rs[256];
    __shared__ float rs2[256];
    const long long base = (long long)blockIdx.x * n;
    const int t = threadIdx.x;

    float s = 0.0f, s2 = 0.0f;
    for (int i = t; i < n; i += 256) {
        float x = a[base + i] + b[base + i];
        s += x; s2 += x * x;
    }
    rs[t] = s; rs2[t] = s2;
    __syncthreads();
    for (int off = 128; off > 0; off >>= 1) {
        if (t < off) { rs[t] += rs[t + off]; rs2[t] += rs2[t + off]; }
        __syncthreads();
    }
    const float invN = 1.0f / (float)n;
    const float mean = rs[0] * invN;
    const float var  = rs2[0] * invN - mean * mean;
    const float inv  = rsqrtf(var + 1e-5f);
    for (int i = t; i < n; i += 256) {
        float x = a[base + i] + b[base + i];
        out[base + i] = (x - mean) * inv * gamma[i] + beta[i];
    }
}

// ---------------------------------------------------------------------------
extern "C" void kernel_launch(void* const* d_in, const int* in_sizes, int n_in,
                              void* d_out, int out_size, void* d_ws, size_t ws_size,
                              hipStream_t stream)
{
    (void)in_sizes; (void)n_in; (void)out_size; (void)ws_size;

    const float* X_in  = (const float*)d_in[0]; // [B,C,N]
    const float* T     = (const float*)d_in[1]; // [B,C,L]
    const float* Wsrc[5] = { (const float*)d_in[2], (const float*)d_in[3],
                             (const float*)d_in[4], (const float*)d_in[5],
                             (const float*)d_in[6] };               // Wq Wk Wv Wf1 Wf2
    const float* gamma = (const float*)d_in[7];
    const float* beta  = (const float*)d_in[8];
    float* out = (float*)d_out;

    const int B = 64, C = 1024, N = 128, L = 512;
    const long long CN = (long long)C * N;   // 131072
    const long long CL = (long long)C * L;   // 524288
    const long long NC = (long long)N * C;
    const long long LC = (long long)L * C;
    const long long NL = (long long)N * L;   // 65536

    // ---- workspace carving (bytes), with aliasing of dead buffers ----
    char* W = (char*)d_ws;
    size_t o = 0;
    auto carve = [&](size_t bytes) { void* p = W + o; o += (bytes + 255) & ~(size_t)255; return p; };
    unsigned short* Wb[5];
    for (int i = 0; i < 5; ++i) Wb[i] = (unsigned short*)carve((size_t)C * C * 2);
    unsigned short* XinT = (unsigned short*)carve((size_t)B * NC * 2); // [B][N,C]
    unsigned short* Tt   = (unsigned short*)carve((size_t)B * LC * 2); // [B][L,C]
    unsigned short* qT   = (unsigned short*)carve((size_t)B * NC * 2); // [B][N,C]
    unsigned short* kT   = (unsigned short*)carve((size_t)B * LC * 2); // [B][L,C]
    unsigned short* vbf  = (unsigned short*)carve((size_t)B * CL * 2); // [B][C,L]
    unsigned short* attn = (unsigned short*)carve((size_t)B * NL * 2); // [B][N,L]
    float* ctx  = (float*)carve((size_t)B * CN * 4);                   // [B][C,N]
    float* Xout = (float*)carve((size_t)B * CN * 4);                   // [B][C,N]
    // aliases into dead regions:
    float*          sim   = (float*)Tt;                                // 16MB in Tt (dead after v GEMM)
    unsigned short* XoutT = (unsigned short*)((char*)Tt + (1u << 25)); // Tt+32MB, disjoint from sim
    unsigned short* hT    = qT;                                        // qT dead after sim GEMM
    float*          ffn   = ctx;                                       // ctx dead after LN1

    dim3 blk(256);

    // ---- precision prep: bf16 weights, transposed bf16 activations ----
    for (int i = 0; i < 5; ++i)
        f32_to_bf16<<<dim3((C * C) / 1024), blk, 0, stream>>>(Wsrc[i], Wb[i]);
    transpose_f32_to_bf16<<<dim3(32, 4, 64),  blk, 0, stream>>>(X_in, XinT, C, N);
    transpose_f32_to_bf16<<<dim3(32, 16, 64), blk, 0, stream>>>(T,    Tt,   C, L);

    // ---- q^T = (Wq X)^T ----
    gemm_bf16_wmma<1, 0><<<dim3(8, 1, 64), blk, 0, stream>>>(
        Wb[0], XinT, qT, C, N, C, C, C, C, 0, NC, NC, 1.0f);
    // ---- k^T = (Wk T)^T ----
    gemm_bf16_wmma<1, 0><<<dim3(8, 4, 64), blk, 0, stream>>>(
        Wb[1], Tt, kT, C, L, C, C, C, C, 0, LC, LC, 1.0f);
    // ---- v = Wv T (bf16, natural layout) ----
    gemm_bf16_wmma<2, 0><<<dim3(8, 4, 64), blk, 0, stream>>>(
        Wb[2], Tt, vbf, C, L, C, C, C, L, 0, LC, CL, 1.0f);
    // ---- sim = (q^T k) / sqrt(C)  : A=qT [N,C], B=kT [L,C] -> f32 [N,L] ----
    gemm_bf16_wmma<0, 0><<<dim3(1, 4, 64), blk, 0, stream>>>(
        qT, kT, sim, N, L, C, C, C, L, NC, LC, NL, 0.03125f);
    // ---- attn = softmax(sim) -> bf16 [N,L] (B-layout for ctx GEMM) ----
    softmax_rows_512<<<dim3((B * N) / 8), blk, 0, stream>>>(sim, attn);
    // ---- ctx = v attn^T : A=v [C,L], B=attn [N,L] -> f32 [C,N] ----
    gemm_bf16_wmma<0, 0><<<dim3(8, 1, 64), blk, 0, stream>>>(
        vbf, attn, ctx, C, N, L, L, L, N, CL, NL, CN, 1.0f);
    // ---- X_out = LN(X_in + ctx), plus transposed bf16 copy for next GEMM ----
    layernorm_add<<<dim3(B), blk, 0, stream>>>(X_in, ctx, gamma, beta, Xout, (int)CN);
    transpose_f32_to_bf16<<<dim3(32, 4, 64), blk, 0, stream>>>(Xout, XoutT, C, N);
    // ---- h^T = relu(Wf1 X_out)^T ----
    gemm_bf16_wmma<1, 1><<<dim3(8, 1, 64), blk, 0, stream>>>(
        Wb[3], XoutT, hT, C, N, C, C, C, C, 0, NC, NC, 1.0f);
    // ---- ffn = Wf2 h : f32 [C,N] ----
    gemm_bf16_wmma<0, 0><<<dim3(8, 1, 64), blk, 0, stream>>>(
        Wb[4], hT, ffn, C, N, C, C, C, N, 0, NC, CN, 1.0f);
    // ---- out = LN(X_out + ffn) ----
    layernorm_add<<<dim3(B), blk, 0, stream>>>(Xout, ffn, gamma, beta, out, (int)CN);
}
